// Dynamic_conv2d_71012989272361
// MI455X (gfx1250) — matrix operand order repaired and checked by reading, not compiled, not run on hardware
//
#include <hip/hip_runtime.h>
#include <hip/hip_bf16.h>
#include <math.h>

typedef __attribute__((ext_vector_type(16))) _Float16 v16h;
typedef __attribute__((ext_vector_type(8)))  float    v8f;

#define BATCH 16
#define CH    64
#define HID   17
#define KEXP  4
#define HW    256
#define TEMPERATURE 30.0f
#define BN_EPS 1e-5f

// ---- workspace layout (bytes) ----
#define WS_POOLED 0        // [16][64] f32   (4096 B)
#define WS_SIG    4096     // [16][64] f32   (4096 B)
#define WS_SATT   8192     // [16][4]  f32   (256 B)
#define WS_AGGW   8448     // [16][64][9] f32 (36864 B)
#define WS_WHALF  45312    // [64][64] f16   (8192 B)

// ---------------------------------------------------------------------------
// Kernel 1: global average pool  x[b,c,:,:] -> pooled[b,c]
// ---------------------------------------------------------------------------
__global__ void pool_kernel(const float* __restrict__ x, float* __restrict__ pooled) {
    const int c = blockIdx.x;
    const int b = blockIdx.y;
    const float4* p = (const float4*)(x + (((size_t)b * CH + c) << 16));
    float s = 0.0f;
    for (int i = threadIdx.x; i < (HW * HW / 4); i += 256) {
        float4 v = p[i];
        s += v.x + v.y + v.z + v.w;
    }
    __shared__ float red[256];
    red[threadIdx.x] = s;
    __syncthreads();
    for (int off = 128; off > 0; off >>= 1) {
        if ((int)threadIdx.x < off) red[threadIdx.x] += red[threadIdx.x + off];
        __syncthreads();
    }
    if (threadIdx.x == 0)
        pooled[b * CH + c] = red[0] * (1.0f / (HW * HW));
}

// ---------------------------------------------------------------------------
// Kernel 2: attention MLP + softmax + sigmoid gate + kernel aggregation +
//           conv1 weight -> f16.  One block of 256 threads.
// ---------------------------------------------------------------------------
__global__ void attn_kernel(const float* __restrict__ pooled,
                            const float* __restrict__ att_w1,
                            const float* __restrict__ bn_gamma,
                            const float* __restrict__ bn_beta,
                            const float* __restrict__ att_w2,
                            const float* __restrict__ weight,   // [4][64][1][3][3]
                            const float* __restrict__ conv1_w,  // [64][64]
                            float* __restrict__ sig,
                            float* __restrict__ satt,
                            float* __restrict__ aggw,
                            _Float16* __restrict__ whalf) {
    const int t = threadIdx.x;

    // --- phase 1: softmax attention, one thread per batch sample ---
    if (t < BATCH) {
        float a[HID];
        for (int h = 0; h < HID; ++h) {
            float acc = 0.0f;
            for (int c = 0; c < CH; ++c)
                acc += pooled[t * CH + c] * att_w1[h * CH + c];
            float scale = bn_gamma[h] * rsqrtf(1.0f + BN_EPS);
            acc = acc * scale + bn_beta[h];
            a[h] = acc > 0.0f ? acc : 0.0f;
        }
        float logits[KEXP];
        float m = -1e30f;
        for (int k = 0; k < KEXP; ++k) {
            float acc = 0.0f;
            for (int h = 0; h < HID; ++h)
                acc += a[h] * att_w2[k * HID + h];
            logits[k] = acc / TEMPERATURE;
            m = fmaxf(m, logits[k]);
        }
        float esum = 0.0f;
        float e[KEXP];
        for (int k = 0; k < KEXP; ++k) { e[k] = expf(logits[k] - m); esum += e[k]; }
        for (int k = 0; k < KEXP; ++k) satt[t * KEXP + k] = e[k] / esum;
    }
    __syncthreads();

    // --- phase 2: sigmoid gate ---
    for (int i = t; i < BATCH * CH; i += 256)
        sig[i] = 1.0f / (1.0f + expf(-pooled[i]));

    // --- phase 3: per-sample aggregated depthwise kernels ---
    for (int i = t; i < BATCH * CH * 9; i += 256) {
        int b = i / (CH * 9);
        int rem = i - b * (CH * 9);
        int c = rem / 9;
        int j = rem - c * 9;
        float acc = 0.0f;
        for (int k = 0; k < KEXP; ++k)
            acc += satt[b * KEXP + k] * weight[k * (CH * 9) + c * 9 + j];
        aggw[i] = acc;
    }

    // --- phase 4: conv1 weights to f16 for WMMA ---
    for (int i = t; i < CH * CH; i += 256)
        whalf[i] = (_Float16)conv1_w[i];
}

// ---------------------------------------------------------------------------
// Kernel 3: fused gate + depthwise 3x3 + 1x1 conv (WMMA) for one
//           [2 rows x 32 cols] tile of one batch sample, all 64 channels.
//           256 threads = 8 waves.
// ---------------------------------------------------------------------------
__global__ void __launch_bounds__(256, 2)
dynconv_main_kernel(const float* __restrict__ x,
                    const float* __restrict__ sig,
                    const float* __restrict__ aggw,
                    const _Float16* __restrict__ whalf,
                    const float* __restrict__ bias,
                    float* __restrict__ outp) {
    __shared__ float    xs[CH][4][34];   // gated input halo: 4 rows x 34 cols
    __shared__ _Float16 Yl[64][68];      // depthwise out, [pixel][channel], padded

    const int b  = blockIdx.z;
    const int h0 = blockIdx.y * 2;       // 2 output rows per block
    const int w0 = blockIdx.x * 32;      // 32 output cols per block
    const int tid = threadIdx.x;

    // ---- stage 1: load gated halo into LDS (coalesced over cols) ----
    for (int idx = tid; idx < CH * 4 * 34; idx += 256) {
        int c   = idx / 136;
        int rem = idx - c * 136;
        int r   = rem / 34;
        int col = rem - r * 34;
        int gh = h0 - 1 + r;
        int gw = w0 - 1 + col;
        float v = 0.0f;
        if ((unsigned)gh < (unsigned)HW && (unsigned)gw < (unsigned)HW)
            v = x[(((size_t)b * CH + c) * HW + gh) * HW + gw] * sig[b * CH + c];
        xs[c][r][col] = v;
    }
    __syncthreads();

    // ---- stage 2: depthwise 3x3; each thread: one channel, 16 pixels ----
    {
        const int c = tid >> 2;
        const int q = tid & 3;
        float w9[9];
#pragma unroll
        for (int j = 0; j < 9; ++j) w9[j] = aggw[(b * CH + c) * 9 + j];
        const int row     = q >> 1;
        const int colbase = (q & 1) * 16;
#pragma unroll
        for (int i = 0; i < 16; ++i) {
            int col = colbase + i;
            float acc = 0.0f;
#pragma unroll
            for (int dr = 0; dr < 3; ++dr)
#pragma unroll
                for (int dc = 0; dc < 3; ++dc)
                    acc = fmaf(xs[c][row + dr][col + dc], w9[dr * 3 + dc], acc);
            Yl[row * 32 + col][c] = (_Float16)acc;
        }
    }
    __syncthreads();

    // ---- stage 3: 1x1 conv as 64x64 @ 64x64 GEMM via WMMA f16->f32 ----
    // 8 waves; wave w: output rows o in [16*(w&3), +16);
    // waves 0-3 cover pixel row 0 (p-tiles 0,1), waves 4-7 pixel row 1.
    const int lane = tid & 31;
    const int wv   = tid >> 5;
    const int lh   = lane & 15;
    const int sel  = lane >> 4;
    const int o0   = (wv & 3) * 16;
    const int ptb  = (wv >> 2) * 2;      // 0 or 2
    const int prow = ptb >> 1;           // output row within tile (0 or 1)

    // A fragments: conv1 weights [o][c], per ISA 16x32 f16 A layout
    v16h afrag[2];
#pragma unroll
    for (int kk = 0; kk < 2; ++kk) {
        v16h a;
#pragma unroll
        for (int g = 0; g < 8; ++g) {
            int kbase = ((g & 3) << 1) + ((g >> 2) << 4) + (sel << 3);
            int c = (kk << 5) + kbase;
            a[2 * g]     = whalf[(o0 + lh) * CH + c];
            a[2 * g + 1] = whalf[(o0 + lh) * CH + c + 1];
        }
        afrag[kk] = a;
    }

    // bias for this lane's 8 output channels (contiguous 32B -> vector load)
    float bias8[8];
#pragma unroll
    for (int r = 0; r < 8; ++r)
        bias8[r] = bias[o0 + sel * 8 + r];

    // one base pointer per lane; all stores use immediate offsets
    // (r step = 256KB, p-tile step = 64B; max imm 1.75MB < 8MB limit)
    float* const pbase = outp +
        (((size_t)(b * CH + o0 + sel * 8)) << 16) +
        (size_t)(h0 + prow) * HW + w0 + lh;

#pragma unroll
    for (int pt2 = 0; pt2 < 2; ++pt2) {
        const int p = (ptb + pt2) * 16 + lh;
        v8f acc;
#pragma unroll
        for (int r = 0; r < 8; ++r) acc[r] = bias8[r];
#pragma unroll
        for (int kk = 0; kk < 2; ++kk) {
            v16h bfr;
#pragma unroll
            for (int g = 0; g < 8; ++g) {
                int c = (kk << 5) + ((g & 3) << 1) + ((g >> 2) << 4) + (sel << 3);
                bfr[2 * g]     = Yl[p][c];
                bfr[2 * g + 1] = Yl[p][c + 1];
            }
            acc = __builtin_amdgcn_wmma_f32_16x16x32_f16(
                false, afrag[kk], false, bfr, (short)0, acc, false, false);
        }
        // D layout: VGPR r -> M = r + sel*8, N = lh
#pragma unroll
        for (int r = 0; r < 8; ++r)
            __builtin_nontemporal_store(acc[r], pbase + ((size_t)r << 16) + pt2 * 16);
    }
}

// ---------------------------------------------------------------------------
extern "C" void kernel_launch(void* const* d_in, const int* in_sizes, int n_in,
                              void* d_out, int out_size, void* d_ws, size_t ws_size,
                              hipStream_t stream) {
    (void)in_sizes; (void)n_in; (void)out_size; (void)ws_size;

    const float* x        = (const float*)d_in[0];
    const float* att_w1   = (const float*)d_in[1];
    const float* bn_gamma = (const float*)d_in[2];
    const float* bn_beta  = (const float*)d_in[3];
    const float* att_w2   = (const float*)d_in[4];
    const float* weight   = (const float*)d_in[5];
    const float* conv1_w  = (const float*)d_in[6];
    const float* conv1_b  = (const float*)d_in[7];
    float* outp = (float*)d_out;

    char* ws = (char*)d_ws;
    float*    pooled = (float*)(ws + WS_POOLED);
    float*    sig    = (float*)(ws + WS_SIG);
    float*    satt   = (float*)(ws + WS_SATT);
    float*    aggw   = (float*)(ws + WS_AGGW);
    _Float16* whalf  = (_Float16*)(ws + WS_WHALF);

    // 1) global average pool
    pool_kernel<<<dim3(CH, BATCH, 1), 256, 0, stream>>>(x, pooled);

    // 2) attention / gates / aggregated kernels / f16 weights
    attn_kernel<<<1, 256, 0, stream>>>(pooled, att_w1, bn_gamma, bn_beta,
                                       att_w2, weight, conv1_w,
                                       sig, satt, aggw, whalf);

    // 3) fused gate + depthwise + 1x1 conv (WMMA)
    dynconv_main_kernel<<<dim3(HW / 32, HW / 2, BATCH), 256, 0, stream>>>(
        x, sig, aggw, whalf, conv1_b, outp);
}
